// GRUCell_25391846654706
// MI455X (gfx1250) — compile-verified, matrix operand order checked by reading, not run on hardware
//
#include <hip/hip_runtime.h>
#include <hip/hip_bf16.h>

// ---------- CDNA5 WMMA types ----------
typedef __attribute__((ext_vector_type(16))) __bf16   bf16x16;
typedef __attribute__((ext_vector_type(8)))  float    f32x8;
typedef __attribute__((ext_vector_type(4)))  float    f32x4;
typedef __attribute__((ext_vector_type(4)))  unsigned int u32x4;

#define DIM   1024
#define TWOD  2048
#define BATCH 16384

// float -> bf16 (round to nearest even)
__device__ __forceinline__ unsigned short f32_to_bf16(float f) {
    unsigned int u = __float_as_uint(f);
    u += 0x7FFFu + ((u >> 16) & 1u);
    return (unsigned short)(u >> 16);
}

// fast activations: v_exp_f32 + v_rcp_f32
__device__ __forceinline__ float fast_sigmoid(float g) {
    const float e = __expf(-g);
    return __builtin_amdgcn_rcpf(1.0f + e);
}
__device__ __forceinline__ float fast_tanh(float g) {
    const float e = __expf(-2.0f * g);
    return (1.0f - e) * __builtin_amdgcn_rcpf(1.0f + e);
}

struct Frag32B { u32x4 lo; u32x4 hi; };

// Load a 16x32 bf16 fragment from a K-major matrix, row stride 2048 elements.
// Lane L<16  : row L, k = k0+{0..7} and k0+{16..23}
// Lane L>=16 : row L-16, k = k0+{8..15} and k0+{24..31}
__device__ __forceinline__ bf16x16 load_frag(const unsigned short* __restrict__ base,
                                             int k0, int lane) {
    const int lr = lane & 15;
    const int hs = (lane >> 4) & 1;
    const unsigned short* p = base + (size_t)lr * TWOD + k0 + hs * 8;
    Frag32B f;
    f.lo = *reinterpret_cast<const u32x4*>(p);
    f.hi = *reinterpret_cast<const u32x4*>(p + 16);
    return __builtin_bit_cast(bf16x16, f);
}

// ---------- Kernel 1: transpose + convert W[K][N] f32 -> Wt[N][K] bf16 ----------
__global__ __launch_bounds__(256) void transpose_to_bf16(
    const float* __restrict__ W, unsigned short* __restrict__ Wt, int K, int N) {
    __shared__ float tile[32][33];
    const int n0 = blockIdx.x * 32;
    const int k0 = blockIdx.y * 32;
    const int tx = threadIdx.x;   // 0..31
    const int ty = threadIdx.y;   // 0..7
#pragma unroll
    for (int i = 0; i < 32; i += 8)
        tile[ty + i][tx] = W[(size_t)(k0 + ty + i) * N + (n0 + tx)];
    __syncthreads();
#pragma unroll
    for (int i = 0; i < 32; i += 8)
        Wt[(size_t)(n0 + ty + i) * K + (k0 + tx)] = f32_to_bf16(tile[tx][ty + i]);
}

// ---------- Kernel 2/4: layernorm(concat(x, h[*r])) -> bf16 [row, 2048] ----------
// 256 threads/row, 8 CONTIGUOUS elements per thread:
//   2x b128 float4 loads + 1x b128 packed-bf16 store per thread.
// For MUL_R, r comes from the gates buffer gbuf[row][0:1024] (row stride 2048).
template <bool MUL_R>
__global__ __launch_bounds__(256) void ln_to_bf16(
    const float* __restrict__ x, const float* __restrict__ h,
    const float* __restrict__ rg,     // gates buffer, row stride 2048 (MUL_R only)
    const float* __restrict__ gamma, const float* __restrict__ beta,
    const float* __restrict__ mask,
    unsigned short* __restrict__ A) {
    const int row = blockIdx.x;
    const int tid = threadIdx.x;
    const int c0  = tid * 8;                  // 8 contiguous columns in [0,2048)

    f32x4 va, vb;
    if (tid < 128) {                          // uniform branch: from x
        const f32x4* p = reinterpret_cast<const f32x4*>(x + (size_t)row * DIM + c0);
        va = p[0]; vb = p[1];
    } else {                                  // from h (optionally * r)
        const int ch = c0 - DIM;
        const f32x4* p = reinterpret_cast<const f32x4*>(h + (size_t)row * DIM + ch);
        va = p[0]; vb = p[1];
        if (MUL_R) {
            const f32x4* pr = reinterpret_cast<const f32x4*>(rg + (size_t)row * TWOD + ch);
            f32x4 ra = pr[0], rb = pr[1];
#pragma unroll
            for (int j = 0; j < 4; ++j) { va[j] *= ra[j]; vb[j] *= rb[j]; }
        }
    }

    float v[8];
#pragma unroll
    for (int j = 0; j < 4; ++j) { v[j] = va[j]; v[4 + j] = vb[j]; }

    float s = 0.f, s2 = 0.f;
#pragma unroll
    for (int j = 0; j < 8; ++j) { s += v[j]; s2 += v[j] * v[j]; }

    // wave32 butterfly reduction
#pragma unroll
    for (int off = 16; off > 0; off >>= 1) {
        s  += __shfl_xor(s,  off, 32);
        s2 += __shfl_xor(s2, off, 32);
    }
    __shared__ float sh[16];
    const int wave = tid >> 5, lane = tid & 31;
    if (lane == 0) { sh[wave] = s; sh[8 + wave] = s2; }
    __syncthreads();
    s = 0.f; s2 = 0.f;
#pragma unroll
    for (int w = 0; w < 8; ++w) { s += sh[w]; s2 += sh[8 + w]; }

    const float mu   = s * (1.0f / TWOD);
    const float var  = s2 * (1.0f / TWOD) - mu * mu;
    const float rstd = rsqrtf(var + 1e-5f);

    const f32x4* gm = reinterpret_cast<const f32x4*>(gamma + c0);
    const f32x4* bt = reinterpret_cast<const f32x4*>(beta + c0);
    const f32x4* mk = reinterpret_cast<const f32x4*>(mask + c0);
    f32x4 g0 = gm[0], g1v = gm[1], b0 = bt[0], b1v = bt[1], m0 = mk[0], m1v = mk[1];

    u32x4 packed;
#pragma unroll
    for (int q = 0; q < 4; ++q) {
        const int j0 = 2 * q, j1 = 2 * q + 1;
        const float ga = (j0 < 4) ? g0[j0] : g1v[j0 - 4];
        const float gb = (j1 < 4) ? g0[j1] : g1v[j1 - 4];
        const float ba = (j0 < 4) ? b0[j0] : b1v[j0 - 4];
        const float bb = (j1 < 4) ? b0[j1] : b1v[j1 - 4];
        const float ma = (j0 < 4) ? m0[j0] : m1v[j0 - 4];
        const float mb = (j1 < 4) ? m0[j1] : m1v[j1 - 4];
        const float oa = ((v[j0] - mu) * rstd * ga + ba) * ma;
        const float ob = ((v[j1] - mu) * rstd * gb + bb) * mb;
        packed[q] = (unsigned int)f32_to_bf16(oa) |
                    ((unsigned int)f32_to_bf16(ob) << 16);
    }
    *reinterpret_cast<u32x4*>(A + (size_t)row * TWOD + c0) = packed;
}

// ---------- Kernel 3/5: bf16 WMMA GEMM with fused epilogue ----------
// A [Btot, 2048] bf16 (K-major), Wt [N, 2048] bf16 (K-major), K = 2048.
// Block: 256 threads = 8 waves arranged 4(M) x 2(N); block tile M=256, N=128.
// Wave tile 64x64 = 4x4 WMMA tiles -> 16 WMMA per 16 b128 loads per K-step.
// EPI == 1 : gbuf[row*2048 + col] = sigmoid(acc + bias)   (no branch)
// EPI == 2 : u = tanh(acc + bias); out = h*z + (1-z)*u, z = gbuf[row*2048+1024+col]
template <int EPI>
__global__ __launch_bounds__(256) void gemm_bf16_wmma(
    const unsigned short* __restrict__ A,
    const unsigned short* __restrict__ Wt,
    const float* __restrict__ bias,
    const float* __restrict__ hin,    // EPI==2
    const float* __restrict__ gbuf,   // EPI==2: gates (z at col+1024), stride 2048
    float* __restrict__ out0) {       // EPI==1: gbuf ; EPI==2: h_new
    const int tid  = threadIdx.x;
    const int lane = tid & 31;
    const int wave = tid >> 5;
    const int wm = wave & 3;          // 4 wave-rows
    const int wn = wave >> 2;         // 2 wave-cols
    const int Mb = blockIdx.y * 256 + wm * 64;
    const int Nb = blockIdx.x * 128 + wn * 64;

    f32x8 acc[4][4];
#pragma unroll
    for (int mt = 0; mt < 4; ++mt)
#pragma unroll
        for (int nt = 0; nt < 4; ++nt) acc[mt][nt] = {};

    const unsigned short* Abase = A  + (size_t)Mb * TWOD;
    const unsigned short* Bbase = Wt + (size_t)Nb * TWOD;

    for (int k0 = 0; k0 < TWOD; k0 += 32) {
        bf16x16 af[4], bfr[4];
#pragma unroll
        for (int mt = 0; mt < 4; ++mt)
            af[mt] = load_frag(Abase + (size_t)mt * 16 * TWOD, k0, lane);
#pragma unroll
        for (int nt = 0; nt < 4; ++nt)
            bfr[nt] = load_frag(Bbase + (size_t)nt * 16 * TWOD, k0, lane);
#pragma unroll
        for (int mt = 0; mt < 4; ++mt)
#pragma unroll
            for (int nt = 0; nt < 4; ++nt)
                acc[mt][nt] = __builtin_amdgcn_wmma_f32_16x16x32_bf16(
                    false, af[mt], false, bfr[nt], (short)0, acc[mt][nt],
                    false, false);
    }

    // C/D layout: VGPR j -> row j + 8*(lane>=16); col = lane&15
    const int lr = lane & 15;
    const int hs = (lane >> 4) & 1;
#pragma unroll
    for (int mt = 0; mt < 4; ++mt) {
#pragma unroll
        for (int nt = 0; nt < 4; ++nt) {
            const int col  = Nb + nt * 16 + lr;
            const float bb = bias[col];
            const int row0 = Mb + mt * 16 + hs * 8;
#pragma unroll
            for (int j = 0; j < 8; ++j) {
                const int row = row0 + j;
                const float g = acc[mt][nt][j] + bb;
                if (EPI == 1) {
                    out0[(size_t)row * TWOD + col] = fast_sigmoid(g);
                } else {
                    const float u = fast_tanh(g);
                    const float z = gbuf[(size_t)row * TWOD + DIM + col];
                    const size_t idx = (size_t)row * DIM + col;
                    out0[idx] = hin[idx] * z + (1.0f - z) * u;
                }
            }
        }
    }
}

extern "C" void kernel_launch(void* const* d_in, const int* in_sizes, int n_in,
                              void* d_out, int out_size, void* d_ws, size_t ws_size,
                              hipStream_t stream) {
    (void)in_sizes; (void)n_in; (void)out_size; (void)ws_size;
    const float* x    = (const float*)d_in[0];
    const float* h    = (const float*)d_in[1];
    const float* Wg   = (const float*)d_in[2];   // [2048, 2048]
    const float* bg   = (const float*)d_in[3];   // [2048]
    const float* Wu   = (const float*)d_in[4];   // [2048, 1024]
    const float* bu   = (const float*)d_in[5];   // [1024]
    const float* g1   = (const float*)d_in[6];
    const float* b1   = (const float*)d_in[7];
    const float* g2   = (const float*)d_in[8];
    const float* b2   = (const float*)d_in[9];
    const float* mask = (const float*)d_in[10];
    float* out = (float*)d_out;

    char* ws = (char*)d_ws;
    unsigned short* Abuf = (unsigned short*)ws; ws += (size_t)BATCH * TWOD * 2;   //  64 MB
    unsigned short* Wgt  = (unsigned short*)ws; ws += (size_t)TWOD * TWOD * 2;    //   8 MB
    unsigned short* Wut  = (unsigned short*)ws; ws += (size_t)DIM  * TWOD * 2;    //   4 MB
    float* gbuf = (float*)ws;                                                     // 128 MB

    dim3 tb(32, 8);
    // Wg [K=2048, N=2048] -> Wgt [2048, 2048]
    transpose_to_bf16<<<dim3(TWOD / 32, TWOD / 32), tb, 0, stream>>>(Wg, Wgt, TWOD, TWOD);
    // Wu [K=2048, N=1024] -> Wut [1024, 2048]
    transpose_to_bf16<<<dim3(DIM / 32, TWOD / 32), tb, 0, stream>>>(Wu, Wut, TWOD, DIM);

    // LN1(concat(x,h)) -> Abuf
    ln_to_bf16<false><<<BATCH, 256, 0, stream>>>(x, h, nullptr, g1, b1, mask, Abuf);

    // gates = sigmoid(LN1 @ Wg + bg) -> gbuf [B, 2048] (r = cols 0..1023, z = 1024..2047)
    gemm_bf16_wmma<1><<<dim3(TWOD / 128, BATCH / 256), 256, 0, stream>>>(
        Abuf, Wgt, bg, nullptr, nullptr, gbuf);

    // LN2(concat(x, h*r)) -> Abuf (reuse)
    ln_to_bf16<true><<<BATCH, 256, 0, stream>>>(x, h, gbuf, g2, b2, mask, Abuf);

    // u = tanh(LN2 @ Wu + bu) ; h_new = h*z + (1-z)*u
    gemm_bf16_wmma<2><<<dim3(DIM / 128, BATCH / 256), 256, 0, stream>>>(
        Abuf, Wut, bu, h, gbuf, out);
}
